// MHA_19361712570613
// MI455X (gfx1250) — compile-verified, matrix operand order checked by reading, not run on hardware
//
#include <hip/hip_runtime.h>
#include <hip/hip_bf16.h>

// ---------------------------------------------------------------------------
// MHA forward for MI455X (gfx1250, wave32, WMMA 16x16x32 f16 -> f32 acc)
//   B=2, S=2048, H=1024, heads=16, head_dim=64
// Pipeline:
//   1) Qh = f16(q @ wq^T + bq)   (WMMA GEMM, f32 in -> f16 out)
//   2) Kh, Vh likewise
//   3) ctx = f16(flash-attention(Qh,Kh,Vh))   (WMMA QK^T and PV;
//        K tile staged by the Tensor Data Mover with LDS row padding)
//   4) out = f32(ctx @ wo^T + bo)             (WMMA GEMM, f16 in -> f32 out)
// ---------------------------------------------------------------------------

typedef __attribute__((ext_vector_type(16))) _Float16 v16h;
typedef __attribute__((ext_vector_type(8)))  float    v8f;
typedef __attribute__((ext_vector_type(4)))  _Float16 half4;
typedef __attribute__((ext_vector_type(4)))  float    f32x4;
typedef __attribute__((ext_vector_type(4)))  unsigned int u32x4;
typedef __attribute__((ext_vector_type(4)))  int      i32x4;
typedef __attribute__((ext_vector_type(8)))  int      i32x8;

union V16H { v16h v; u32x4 q[2]; };

#if defined(__has_builtin)
#if __has_builtin(__builtin_amdgcn_tensor_load_to_lds) && \
    __has_builtin(__builtin_amdgcn_s_wait_tensorcnt)
#define ATTN_USE_TDM 1
#endif
#endif
#ifndef ATTN_USE_TDM
#define ATTN_USE_TDM 0
#endif

static __device__ inline v8f wmma_f16(v16h a, v16h b, v8f c) {
  // 8 args: (neg_a, A, neg_b, B, c_mod, C, reuse_a, reuse_b)
  return __builtin_amdgcn_wmma_f32_16x16x32_f16(false, a, false, b,
                                                (short)0, c, false, false);
}

static __device__ inline half4 ld4cvt(const float* p) {
  return __builtin_convertvector(*(const f32x4*)p, half4);
}
static __device__ inline half4 ld4cvt(const _Float16* p) {
  return *(const half4*)p;
}
static __device__ inline void st1(float* C, size_t i, float v)    { C[i] = v; }
static __device__ inline void st1(_Float16* C, size_t i, float v) { C[i] = (_Float16)v; }

// ---------------------------------------------------------------------------
// GEMM: C[M,N] = A[M,K] @ W[N,K]^T + bias[N]   (torch Linear layout)
// Block: 256 threads (8 waves), tile 128x128, K-step 32.
// LDS tiles are row-major f16 with row stride 40 halves (80B, conflict-free).
// Each lane's WMMA fragment = two 16B LDS loads matching the ISA layout:
//   lane L (m = L%16, h2 = L/16): K chunks [8*h2, 8*h2+7] and [16+8*h2, ...].
// ---------------------------------------------------------------------------
template <typename TA, typename TO>
__global__ __launch_bounds__(256)
void gemm_wmma_bias(const TA* __restrict__ A, const float* __restrict__ W,
                    const float* __restrict__ bias, TO* __restrict__ C,
                    int M, int N, int K) {
  constexpr int BM = 128, BN = 128, BK = 32, LDT = 40;
  __shared__ _Float16 sA[BM * LDT];
  __shared__ _Float16 sB[BN * LDT];

  const int tid    = threadIdx.x;
  const int wave   = tid >> 5;
  const int lane   = tid & 31;
  const int lane16 = lane & 15;
  const int h2     = lane >> 4;
  const int m0     = blockIdx.x * BM;
  const int n0     = blockIdx.y * BN;

  v8f acc[8];
#pragma unroll
  for (int j = 0; j < 8; ++j) acc[j] = (v8f){0,0,0,0,0,0,0,0};

  for (int k0 = 0; k0 < K; k0 += BK) {
    // ---- stage A (128x32) and W (128x32) tiles into LDS as f16 ----
#pragma unroll
    for (int t = 0; t < 4; ++t) {
      const int chunk = tid + t * 256;       // 1024 chunks of 4 elements
      const int row   = chunk >> 3;
      const int c4    = (chunk & 7) * 4;
      *(half4*)&sA[row * LDT + c4] = ld4cvt(&A[(size_t)(m0 + row) * K + k0 + c4]);
      *(half4*)&sB[row * LDT + c4] = ld4cvt(&W[(size_t)(n0 + row) * K + k0 + c4]);
    }
    __syncthreads();

    if (k0 + BK < K) {  // hint: prefetch next K-step (global_prefetch_b8)
      __builtin_prefetch(&A[(size_t)(m0 + tid / 2) * K + k0 + BK], 0, 1);
      __builtin_prefetch(&W[(size_t)(n0 + tid / 2) * K + k0 + BK], 0, 1);
    }

    // ---- per-wave compute: 16 rows x 128 cols (8 WMMA tiles) ----
    V16H av;
    const _Float16* ap = &sA[(wave * 16 + lane16) * LDT];
    av.q[0] = *(const u32x4*)(ap + 8 * h2);
    av.q[1] = *(const u32x4*)(ap + 16 + 8 * h2);
#pragma unroll
    for (int j = 0; j < 8; ++j) {
      V16H bv;
      const _Float16* bp = &sB[(j * 16 + lane16) * LDT];
      bv.q[0] = *(const u32x4*)(bp + 8 * h2);
      bv.q[1] = *(const u32x4*)(bp + 16 + 8 * h2);
      acc[j] = wmma_f16(av.v, bv.v, acc[j]);
    }
    __syncthreads();
  }

  // ---- epilogue: bias add, store (C layout: VGPR r -> row r + 8*h2) ----
#pragma unroll
  for (int j = 0; j < 8; ++j) {
    const int n  = n0 + j * 16 + lane16;
    const float bb = bias[n];
#pragma unroll
    for (int r = 0; r < 8; ++r) {
      const int m = m0 + wave * 16 + r + 8 * h2;
      st1(C, (size_t)m * N + n, acc[j][r] + bb);
    }
  }
}

// ---------------------------------------------------------------------------
// Flash attention: one block = (batch b, head h, 128 q rows); 8 waves,
// each wave owns 16 q rows. Streams keys in tiles of 64.
//   scores = Q K^T / 8 ; online softmax ; O += P V ; O /= l at the end.
// K tile is staged via the Tensor Data Mover (tensor_load_to_lds) with
// pad_enable producing the 72-half LDS row stride directly.
// ---------------------------------------------------------------------------
__global__ __launch_bounds__(256)
void attn_wmma(const _Float16* __restrict__ Qh, const _Float16* __restrict__ Kh,
               const _Float16* __restrict__ Vh, _Float16* __restrict__ ctx) {
  constexpr int S = 2048, HC = 1024, D = 64, KT = 64, LDT = 72;
  __shared__ _Float16 sK[KT * LDT];       // K tile, row = key, cols = d
  __shared__ _Float16 sVt[D * LDT];       // V tile transposed: row = d, cols = key
  __shared__ _Float16 sP[8 * 16 * LDT];   // per-wave P slab (16 x 64)

  const int tid    = threadIdx.x;
  const int wave   = tid >> 5;
  const int lane   = tid & 31;
  const int lane16 = lane & 15;
  const int h2     = lane >> 4;
  const int b      = blockIdx.z;
  const int h      = blockIdx.y;
  const int q0     = blockIdx.x * 128 + wave * 16;

  // ---- load Q fragments once (A-operand layout, straight from global) ----
  V16H aq[2];
#pragma unroll
  for (int kk = 0; kk < 2; ++kk) {
    const _Float16* qp = Qh + (size_t)(b * S + q0 + lane16) * HC + h * D + kk * 32;
    aq[kk].q[0] = *(const u32x4*)(qp + 8 * h2);
    aq[kk].q[1] = *(const u32x4*)(qp + 16 + 8 * h2);
  }

  v8f O[4];
#pragma unroll
  for (int t = 0; t < 4; ++t) O[t] = (v8f){0,0,0,0,0,0,0,0};
  float mrow[8], lrow[8];
#pragma unroll
  for (int r = 0; r < 8; ++r) { mrow[r] = -3.0e38f; lrow[r] = 0.0f; }

  for (int kb = 0; kb < S / KT; ++kb) {
    __syncthreads();

    // ---- stage K tile: Tensor Data Mover (64x64 f16 tile, padded rows) ----
#if ATTN_USE_TDM
    if (wave == 0) {
      const unsigned long long ga =
          (unsigned long long)(const void*)(Kh + (size_t)(b * S + kb * KT) * HC + h * D);
      u32x4 g0;
      g0[0] = 1u;                                      // count=1, user descriptor
      g0[1] = (unsigned)(unsigned long long)(const void*)sK;  // lds_addr
      g0[2] = (unsigned)(ga & 0xFFFFFFFFull);          // global_addr[31:0]
      g0[3] = (unsigned)((ga >> 32) & 0x01FFFFFFull)   // global_addr[56:32]
            | (2u << 30);                              // type = 2 ("image")
      i32x8 g1;
      g1[0] = (1 << 16)      // data_size = 2 bytes
            | (1 << 20)      // pad_enable
            | (4 << 22)      // pad_interval: every 32 DWORDs (128B row)
            | (3 << 25);     // pad_amount: 4 DWORDs (16B) -> 72-half stride
      g1[1] = (int)(((unsigned)HC & 0xFFFFu) << 16);   // tensor_dim0[15:0]
      g1[2] = (int)((((unsigned)HC >> 16) & 0xFFFFu)   // tensor_dim0[31:16]
            | ((4096u & 0xFFFFu) << 16));              // tensor_dim1[15:0]
      g1[3] = (int)(((4096u >> 16) & 0xFFFFu)          // tensor_dim1[31:16]
            | ((unsigned)KT << 16));                   // tile_dim0 = 64
      g1[4] = KT;                                      // tile_dim1 = 64 (tile_dim2=0)
      g1[5] = HC;                                      // tensor_dim0_stride[31:0]
      g1[6] = 0;
      g1[7] = 0;
      const i32x4 z4 = {0, 0, 0, 0};
#if __clang_major__ >= 23
      const i32x8 z8 = {0, 0, 0, 0, 0, 0, 0, 0};
      __builtin_amdgcn_tensor_load_to_lds(g0, g1, z4, z4, z8, 0);
#else
      __builtin_amdgcn_tensor_load_to_lds(g0, g1, z4, z4, 0);
#endif
      __builtin_amdgcn_s_wait_tensorcnt(0);
    }
#else
#pragma unroll
    for (int t = 0; t < 2; ++t) {
      const int chunk = tid + t * 256;            // 512 chunks of 8 halves
      const int row   = chunk >> 3;
      const int c8    = (chunk & 7) * 8;
      *(u32x4*)&sK[row * LDT + c8] =
          *(const u32x4*)(Kh + (size_t)(b * S + kb * KT + row) * HC + h * D + c8);
    }
#endif

    // ---- stage V tile transposed (manual: TDM cannot transpose) ----
#pragma unroll
    for (int t = 0; t < 2; ++t) {
      const int chunk = tid + t * 256;
      const int key = chunk >> 3;
      const int d8  = (chunk & 7) * 8;
      u32x4 vv = *(const u32x4*)(Vh + (size_t)(b * S + kb * KT + key) * HC + h * D + d8);
      _Float16 tmp[8];
      *(u32x4*)tmp = vv;
#pragma unroll
      for (int i = 0; i < 8; ++i) sVt[(d8 + i) * LDT + key] = tmp[i];
    }
    __syncthreads();

    if (kb + 1 < S / KT)  // prefetch next key tile
      __builtin_prefetch(Kh + (size_t)(b * S + (kb + 1) * KT + (tid >> 2)) * HC + h * D, 0, 1);

    // ---- scores: 4 tiles of 16x16 over the 64-key block ----
    v8f s[4];
#pragma unroll
    for (int j = 0; j < 4; ++j) {
      v8f c = (v8f){0,0,0,0,0,0,0,0};
#pragma unroll
      for (int kk = 0; kk < 2; ++kk) {
        V16H bv;
        const _Float16* kp = &sK[(j * 16 + lane16) * LDT + kk * 32];
        bv.q[0] = *(const u32x4*)(kp + 8 * h2);
        bv.q[1] = *(const u32x4*)(kp + 16 + 8 * h2);
        c = wmma_f16(aq[kk].v, bv.v, c);
      }
      s[j] = c * 0.125f;  // 1/sqrt(64)
    }

    // ---- online softmax (row reductions over 16-lane half-waves) ----
    float mnew[8];
#pragma unroll
    for (int r = 0; r < 8; ++r) {
      float mx = fmaxf(fmaxf(s[0][r], s[1][r]), fmaxf(s[2][r], s[3][r]));
#pragma unroll
      for (int msk = 1; msk < 16; msk <<= 1) mx = fmaxf(mx, __shfl_xor(mx, msk, 32));
      mnew[r] = fmaxf(mrow[r], mx);
      const float sc = __expf(mrow[r] - mnew[r]);
      lrow[r] *= sc;
#pragma unroll
      for (int t = 0; t < 4; ++t) O[t][r] = O[t][r] * sc;
      mrow[r] = mnew[r];
    }

    // ---- P = exp(s - m): store to per-wave LDS slab, accumulate row sums ----
    _Float16* pbase = &sP[wave * 16 * LDT];
    float rs[8];
#pragma unroll
    for (int r = 0; r < 8; ++r) rs[r] = 0.0f;
#pragma unroll
    for (int j = 0; j < 4; ++j) {
#pragma unroll
      for (int r = 0; r < 8; ++r) {
        const float p = __expf(s[j][r] - mnew[r]);
        pbase[(r + 8 * h2) * LDT + j * 16 + lane16] = (_Float16)p;
        rs[r] += p;
      }
    }
#pragma unroll
    for (int r = 0; r < 8; ++r) {
#pragma unroll
      for (int msk = 1; msk < 16; msk <<= 1) rs[r] += __shfl_xor(rs[r], msk, 32);
      lrow[r] += rs[r];
    }

    // ---- O += P @ V  (K-dim = 64 keys, two WMMA steps) ----
#pragma unroll
    for (int kk2 = 0; kk2 < 2; ++kk2) {
      V16H apv;
      const _Float16* pp = &pbase[lane16 * LDT + kk2 * 32];
      apv.q[0] = *(const u32x4*)(pp + 8 * h2);
      apv.q[1] = *(const u32x4*)(pp + 16 + 8 * h2);
#pragma unroll
      for (int t = 0; t < 4; ++t) {
        V16H bv;
        const _Float16* vp = &sVt[(t * 16 + lane16) * LDT + kk2 * 32];
        bv.q[0] = *(const u32x4*)(vp + 8 * h2);
        bv.q[1] = *(const u32x4*)(vp + 16 + 8 * h2);
        O[t] = wmma_f16(apv.v, bv.v, O[t]);
      }
    }
  }

  // ---- finalize: O /= l, write ctx (f16) ----
#pragma unroll
  for (int r = 0; r < 8; ++r) {
    const float inv = 1.0f / lrow[r];
    const size_t row = (size_t)(b * S + q0 + r + 8 * h2);
#pragma unroll
    for (int t = 0; t < 4; ++t)
      ctx[row * HC + h * D + t * 16 + lane16] = (_Float16)(O[t][r] * inv);
  }
}

// ---------------------------------------------------------------------------
// Host-side launch
// ---------------------------------------------------------------------------
extern "C" void kernel_launch(void* const* d_in, const int* in_sizes, int n_in,
                              void* d_out, int out_size, void* d_ws, size_t ws_size,
                              hipStream_t stream) {
  (void)in_sizes; (void)n_in; (void)out_size; (void)ws_size;
  const float* q    = (const float*)d_in[0];
  const float* k    = (const float*)d_in[1];
  const float* v    = (const float*)d_in[2];
  const float* wq_w = (const float*)d_in[3];
  const float* wq_b = (const float*)d_in[4];
  const float* wk_w = (const float*)d_in[5];
  const float* wk_b = (const float*)d_in[6];
  const float* wv_w = (const float*)d_in[7];
  const float* wv_b = (const float*)d_in[8];
  const float* wo_w = (const float*)d_in[9];
  const float* wo_b = (const float*)d_in[10];
  float* out = (float*)d_out;

  const int M = 4096;   // B*S
  const int H = 1024;
  const size_t MH = (size_t)M * H;

  _Float16* Qh  = (_Float16*)d_ws;
  _Float16* Kh  = Qh + MH;
  _Float16* Vh  = Kh + MH;
  _Float16* ctx = Vh + MH;   // total f16 workspace: 32 MB

  dim3 gg(M / 128, H / 128);
  gemm_wmma_bias<float, _Float16><<<gg, 256, 0, stream>>>(q, wq_w, wq_b, Qh, M, H, H);
  gemm_wmma_bias<float, _Float16><<<gg, 256, 0, stream>>>(k, wk_w, wk_b, Kh, M, H, H);
  gemm_wmma_bias<float, _Float16><<<gg, 256, 0, stream>>>(v, wv_w, wv_b, Vh, M, H, H);

  dim3 ga(2048 / 128, 16, 2);  // (q tiles, heads, batch)
  attn_wmma<<<ga, 256, 0, stream>>>(Qh, Kh, Vh, ctx);

  gemm_wmma_bias<_Float16, float><<<gg, 256, 0, stream>>>(ctx, wo_w, wo_b, out, M, H, H);
}
